// _RGATEncoder_84456236908852
// MI455X (gfx1250) — matrix operand order repaired and checked by reading, not compiled, not run on hardware
//
#include <hip/hip_runtime.h>

typedef __attribute__((ext_vector_type(16))) _Float16 v16h;
typedef __attribute__((ext_vector_type(8)))  _Float16 v8h;
typedef __attribute__((ext_vector_type(8)))  float    v8f;
typedef __attribute__((ext_vector_type(4)))  int      v4i;

// async-to-LDS builtin pointer types: pointee int4 in AS1 (global) / AS3 (LDS)
typedef __attribute__((address_space(1))) v4i gv4i;
typedef __attribute__((address_space(3))) v4i lv4i;

static constexpr int DIM  = 768;
static constexpr int NREL = 3;
static constexpr int H1   = 4;
static constexpr int H2   = 1;

#if __has_builtin(__builtin_amdgcn_global_load_async_to_lds_b128)
#define HAVE_ASYNC_LDS 1
#else
#define HAVE_ASYNC_LDS 0
#endif

// ---------------- helpers ----------------

__device__ __forceinline__ unsigned f2ord(float f) {
  unsigned u = __float_as_uint(f);
  return (u & 0x80000000u) ? ~u : (u | 0x80000000u);
}
__device__ __forceinline__ float ord2f(unsigned u) {
  return (u & 0x80000000u) ? __uint_as_float(u & 0x7fffffffu)
                           : __uint_as_float(~u);
}

__device__ __forceinline__ void wait_async_all() {
#if HAVE_ASYNC_LDS
#if __has_builtin(__builtin_amdgcn_s_wait_asynccnt)
  __builtin_amdgcn_s_wait_asynccnt(0);
#else
  asm volatile("s_wait_asynccnt 0x0" ::: "memory");
#endif
#endif
}

__global__ void zero_kernel(unsigned* __restrict__ p, size_t n) {
  size_t i = (size_t)blockIdx.x * blockDim.x + threadIdx.x;
  if (i < n) p[i] = 0u;
}

__global__ void cast_f16_kernel(const float* __restrict__ in,
                                _Float16* __restrict__ out, size_t n) {
  size_t i = (size_t)blockIdx.x * blockDim.x + threadIdx.x;
  if (i < n) out[i] = (_Float16)in[i];
}

// W: [R][k][n] f32 -> Wt: [R][n][k] f16 (transpose so WMMA B-frags load contiguously)
__global__ void cast_wt_kernel(const float* __restrict__ W,
                               _Float16* __restrict__ Wt) {
  size_t i = (size_t)blockIdx.x * blockDim.x + threadIdx.x;
  const size_t dd = (size_t)DIM * DIM;
  if (i >= (size_t)NREL * dd) return;
  int r = (int)(i / dd);
  size_t rem = i - (size_t)r * dd;
  int k = (int)(rem / DIM);
  int n = (int)(rem % DIM);
  Wt[((size_t)r * DIM + n) * DIM + k] = (_Float16)W[i];
}

// ---------------- WMMA GEMM: C[r][m][n] = A[m][k] * Bt[r][n][k]^T ----------------
// Block = 8 waves = 128(M) x 64(N) tile; B strip staged into LDS in double-buffered
// 64x128-K chunks via async-to-LDS; each wave computes one 16x64 strip (4 wmma/K-step).

static constexpr int KC      = 128;  // K-chunk staged in LDS
static constexpr int LSTRIDE = 136;  // padded LDS row stride (halves): 272B = 17*16B

__device__ __forceinline__ void stage_B(const _Float16* __restrict__ Bbase,
                                        int K, int k0, _Float16* lbuf, int tid) {
  // copy 64 rows x 128 halves (16B per thread-iter, 1024 x b128 total)
#pragma unroll
  for (int i = 0; i < 4; ++i) {
    int idx = tid + i * 256;   // 0..1023
    int col = idx >> 4;        // 64 B-columns (rows of Bt)
    int sub = idx & 15;        // 16 x 16B segments per row
    const _Float16* g = Bbase + (size_t)col * K + k0 + sub * 8;
    _Float16*       l = lbuf + col * LSTRIDE + sub * 8;
#if HAVE_ASYNC_LDS
    __builtin_amdgcn_global_load_async_to_lds_b128((gv4i*)g, (lv4i*)l, 0, 0);
#else
    *(v8h*)l = *(const v8h*)g;
#endif
  }
}

__global__ void __launch_bounds__(256)
gemm_wmma_kernel(const _Float16* __restrict__ A,   // [M, K]
                 const _Float16* __restrict__ Bt,  // [R, Nn, K]
                 float* __restrict__ C,            // [R, M, Nn]
                 int M, int Nn, int K) {
  __shared__ _Float16 lb[2][64 * LSTRIDE];

  const int tid  = threadIdx.x;
  const int w    = tid >> 5;
  const int lane = tid & 31;
  const int row  = lane & 15;
  const int hi   = lane >> 4;

  const int ntiles  = Nn / 64;
  const int mgroups = M / 128;
  const int per_r   = ntiles * mgroups;
  const int r   = blockIdx.x / per_r;
  const int rem = blockIdx.x - r * per_r;
  const int mg  = rem / ntiles;
  const int nt  = rem - mg * ntiles;
  const int mt  = mg * 8 + w;   // this wave's 16-row M tile

  const _Float16* Ar    = A + (size_t)(mt * 16 + row) * K;
  const _Float16* Bbase = Bt + ((size_t)r * Nn + nt * 64) * K;

  v8f acc0 = {}, acc1 = {}, acc2 = {}, acc3 = {};

  stage_B(Bbase, K, 0, lb[0], tid);
  wait_async_all();
  __syncthreads();

  int cur = 0;
  for (int kc = 0; kc < K; kc += KC) {
    if (kc + KC < K) stage_B(Bbase, K, kc + KC, lb[cur ^ 1], tid);
    const _Float16* lcur = lb[cur];
#pragma unroll
    for (int ks = 0; ks < KC; ks += 32) {
      const int k0 = kc + ks;
      // A fragment: lane group 0 holds K 0..15, group 1 K 16..31
      union { v16h v; v8h h[2]; } ua;
      ua.h[0] = *(const v8h*)(Ar + k0 + hi * 16);
      ua.h[1] = *(const v8h*)(Ar + k0 + hi * 16 + 8);
      __builtin_prefetch(Ar + k0 + 128, 0, 1);
      // B fragments from LDS: lane = N column, hi group selects K 0..15 vs 16..31
      union { v16h v; v8h h[2]; } b0, b1, b2, b3;
      const _Float16* l0 = lcur + ( 0 + row) * LSTRIDE + ks + hi * 16;
      const _Float16* l1 = lcur + (16 + row) * LSTRIDE + ks + hi * 16;
      const _Float16* l2 = lcur + (32 + row) * LSTRIDE + ks + hi * 16;
      const _Float16* l3 = lcur + (48 + row) * LSTRIDE + ks + hi * 16;
      b0.h[0] = *(const v8h*)l0;  b0.h[1] = *(const v8h*)(l0 + 8);
      b1.h[0] = *(const v8h*)l1;  b1.h[1] = *(const v8h*)(l1 + 8);
      b2.h[0] = *(const v8h*)l2;  b2.h[1] = *(const v8h*)(l2 + 8);
      b3.h[0] = *(const v8h*)l3;  b3.h[1] = *(const v8h*)(l3 + 8);

      acc0 = __builtin_amdgcn_wmma_f32_16x16x32_f16(false, ua.v, false, b0.v,
                                                    (short)0, acc0, false, false);
      acc1 = __builtin_amdgcn_wmma_f32_16x16x32_f16(false, ua.v, false, b1.v,
                                                    (short)0, acc1, false, false);
      acc2 = __builtin_amdgcn_wmma_f32_16x16x32_f16(false, ua.v, false, b2.v,
                                                    (short)0, acc2, false, false);
      acc3 = __builtin_amdgcn_wmma_f32_16x16x32_f16(false, ua.v, false, b3.v,
                                                    (short)0, acc3, false, false);
    }
    wait_async_all();
    __syncthreads();
    cur ^= 1;
  }

  // C/D layout: VGPR i -> M = hi*8 + i, N = lane&15 (+ 16*j per sub-tile)
  float* Cb = C + (((size_t)r * M + mt * 16 + hi * 8) * Nn) + nt * 64 + row;
#pragma unroll
  for (int i = 0; i < 8; ++i) {
    float* crow = Cb + (size_t)i * Nn;
    crow[0]  = acc0[i];
    crow[16] = acc1[i];
    crow[32] = acc2[i];
    crow[48] = acc3[i];
  }
}

// ---------------- edge attention logits + segment max ----------------
// one wave32 per edge; alpha = leakyrelu(oi.q + oj.k); atomicMax (ordered-uint) into m

template <int H>
__global__ void __launch_bounds__(256)
edge_alpha_kernel(const float* __restrict__ xr,   // [R, N, D]
                  const int* __restrict__ src, const int* __restrict__ dst,
                  const int* __restrict__ et,
                  const float* __restrict__ q,    // [D, H]
                  const float* __restrict__ kk,   // [D, H]
                  float* __restrict__ alpha,      // [E, H]
                  unsigned* __restrict__ m_u,     // [N, H]
                  int E, int Nn) {
  const int e    = blockIdx.x * (blockDim.x >> 5) + (threadIdx.x >> 5);
  const int lane = threadIdx.x & 31;
  if (e >= E) return;
  const int s = src[e], d = dst[e], r = et[e];
  const float* oi = xr + ((size_t)r * Nn + d) * DIM;
  const float* oj = xr + ((size_t)r * Nn + s) * DIM;

  float acc[H];
#pragma unroll
  for (int h = 0; h < H; ++h) acc[h] = 0.f;

  for (int f = lane; f < DIM; f += 32) {
    float vi = oi[f], vj = oj[f];
#pragma unroll
    for (int h = 0; h < H; ++h) acc[h] += vi * q[f * H + h] + vj * kk[f * H + h];
  }
#pragma unroll
  for (int h = 0; h < H; ++h) {
    float t = acc[h];
    for (int off = 16; off; off >>= 1) t += __shfl_xor(t, off, 32);
    if (lane == h) {
      float a = (t > 0.f) ? t : 0.2f * t;     // leaky_relu(0.2)
      alpha[(size_t)e * H + h] = a;
      atomicMax(&m_u[(size_t)d * H + h], f2ord(a));
    }
  }
}

// ---------------- exp(alpha - m) and segment sum ----------------

__global__ void edge_exp_kernel(const int* __restrict__ dst,
                                float* __restrict__ alpha,     // in: logit, out: exp
                                const unsigned* __restrict__ m_u,
                                float* __restrict__ z,
                                int E, int H) {
  int i = blockIdx.x * blockDim.x + threadIdx.x;
  if (i >= E * H) return;
  int e = i / H, h = i - e * H;
  int d = dst[e];
  float ex = __expf(alpha[i] - ord2f(m_u[(size_t)d * H + h]));
  alpha[i] = ex;
  atomicAdd(&z[(size_t)d * H + h], ex);
}

// ---------------- weighted message scatter ----------------
// one wave32 per edge: out[dst] += (e/z) * oj

template <int H>
__global__ void __launch_bounds__(256)
edge_msg_kernel(const float* __restrict__ xr,
                const int* __restrict__ src, const int* __restrict__ dst,
                const int* __restrict__ et,
                const float* __restrict__ ew,    // [E, H] exp weights
                const float* __restrict__ z,     // [N, H]
                float* __restrict__ out,         // [N, D] accumulated
                int E, int Nn) {
  const int e    = blockIdx.x * (blockDim.x >> 5) + (threadIdx.x >> 5);
  const int lane = threadIdx.x & 31;
  if (e >= E) return;
  const int s = src[e], d = dst[e], r = et[e];
  const float* oj = xr + ((size_t)r * Nn + s) * DIM;
  constexpr int O = DIM / H;

  float a[H];
#pragma unroll
  for (int h = 0; h < H; ++h)
    a[h] = ew[(size_t)e * H + h] / (z[(size_t)d * H + h] + 1e-16f);

  float* op = out + (size_t)d * DIM;
  for (int f = lane; f < DIM; f += 32)
    atomicAdd(&op[f], a[f / O] * oj[f]);
}

// ---------------- bias (+ optional ELU) ----------------

__global__ void bias_act_kernel(float* __restrict__ buf, const float* __restrict__ b,
                                size_t n, int do_elu) {
  size_t i = (size_t)blockIdx.x * blockDim.x + threadIdx.x;
  if (i >= n) return;
  int f = (int)(i % DIM);
  float v = buf[i] + b[f];
  if (do_elu) v = (v > 0.f) ? v : (__expf(v) - 1.f);
  buf[i] = v;
}

// ---------------- launcher ----------------

extern "C" void kernel_launch(void* const* d_in, const int* in_sizes, int n_in,
                              void* d_out, int out_size, void* d_ws, size_t ws_size,
                              hipStream_t stream) {
  const float* x  = (const float*)d_in[0];
  const int*   ei = (const int*)d_in[1];
  const int*   et = (const int*)d_in[2];
  const float* W1 = (const float*)d_in[3];
  const float* q1 = (const float*)d_in[4];
  const float* k1 = (const float*)d_in[5];
  const float* b1 = (const float*)d_in[6];
  const float* W2 = (const float*)d_in[7];
  const float* q2 = (const float*)d_in[8];
  const float* k2 = (const float*)d_in[9];
  const float* b2 = (const float*)d_in[10];

  const int Nn = in_sizes[0] / DIM;   // 32768
  const int E  = in_sizes[2];         // 160000
  const int* src = ei;
  const int* dst = ei + E;

  float* out = (float*)d_out;

  // workspace carve (256B aligned)
  size_t off = 0;
  auto carve = [&](size_t bytes) -> void* {
    void* p = (char*)d_ws + off;
    off += (bytes + 255) & ~(size_t)255;
    return p;
  };
  _Float16* xh  = (_Float16*)carve((size_t)Nn * DIM * 2);
  _Float16* Wt  = (_Float16*)carve((size_t)NREL * DIM * DIM * 2);
  float*    xr  = (float*)   carve((size_t)NREL * Nn * DIM * 4);
  float*    alp = (float*)   carve((size_t)E * H1 * 4);
  unsigned* m_u = (unsigned*)carve((size_t)Nn * H1 * 4);
  float*    z   = (float*)   carve((size_t)Nn * H1 * 4);
  float*    hb  = (float*)   carve((size_t)Nn * DIM * 4);
  (void)ws_size; (void)n_in; (void)out_size;

  const int T = 256;
  auto nblk = [&](size_t n) { return (unsigned)((n + T - 1) / T); };
  const size_t nd  = (size_t)Nn * DIM;
  const size_t rdd = (size_t)NREL * DIM * DIM;
  const unsigned gemm_blocks = (unsigned)(NREL * (DIM / 64) * (Nn / 128));
  const unsigned edge_blocks = (unsigned)((E + 7) / 8);  // 8 waves per block

  // ---------- layer 1 ----------
  cast_f16_kernel<<<nblk(nd), T, 0, stream>>>(x, xh, nd);
  cast_wt_kernel<<<nblk(rdd), T, 0, stream>>>(W1, Wt);
  gemm_wmma_kernel<<<gemm_blocks, T, 0, stream>>>(xh, Wt, xr, Nn, DIM, DIM);

  zero_kernel<<<nblk((size_t)Nn * H1), T, 0, stream>>>(m_u, (size_t)Nn * H1);
  zero_kernel<<<nblk((size_t)Nn * H1), T, 0, stream>>>((unsigned*)z, (size_t)Nn * H1);
  zero_kernel<<<nblk(nd), T, 0, stream>>>((unsigned*)hb, nd);

  edge_alpha_kernel<H1><<<edge_blocks, T, 0, stream>>>(xr, src, dst, et, q1, k1,
                                                       alp, m_u, E, Nn);
  edge_exp_kernel<<<nblk((size_t)E * H1), T, 0, stream>>>(dst, alp, m_u, z, E, H1);
  edge_msg_kernel<H1><<<edge_blocks, T, 0, stream>>>(xr, src, dst, et, alp, z, hb, E, Nn);
  bias_act_kernel<<<nblk(nd), T, 0, stream>>>(hb, b1, nd, /*elu=*/1);

  // ---------- layer 2 ----------
  cast_f16_kernel<<<nblk(nd), T, 0, stream>>>(hb, xh, nd);
  cast_wt_kernel<<<nblk(rdd), T, 0, stream>>>(W2, Wt);
  gemm_wmma_kernel<<<gemm_blocks, T, 0, stream>>>(xh, Wt, xr, Nn, DIM, DIM);

  zero_kernel<<<nblk((size_t)Nn * H2), T, 0, stream>>>(m_u, (size_t)Nn * H2);
  zero_kernel<<<nblk((size_t)Nn * H2), T, 0, stream>>>((unsigned*)z, (size_t)Nn * H2);
  zero_kernel<<<nblk(nd), T, 0, stream>>>((unsigned*)out, nd);

  edge_alpha_kernel<H2><<<edge_blocks, T, 0, stream>>>(xr, src, dst, et, q2, k2,
                                                       alp, m_u, E, Nn);
  edge_exp_kernel<<<nblk((size_t)E * H2), T, 0, stream>>>(dst, alp, m_u, z, E, H2);
  edge_msg_kernel<H2><<<edge_blocks, T, 0, stream>>>(xr, src, dst, et, alp, z, out, E, Nn);
  bias_act_kernel<<<nblk(nd), T, 0, stream>>>(out, b2, nd, /*elu=*/0);
}
